// SASRec_CAPE_558345748558
// MI455X (gfx1250) — compile-verified, matrix operand order checked by reading, not run on hardware
//
#include <hip/hip_runtime.h>

// ---------------------------------------------------------------------------
// SASRec + CAPE forward on MI455X (gfx1250, wave32, WMMA).
//
// All GEMMs run on v_wmma_f32_16x16x32_bf16 (bf16 operands, f32 accumulate).
// Activations/intermediates stay fp32 in global memory; bf16 rounding happens
// only on the WMMA operand path (staged through LDS). Fragment loads are two
// ds_load_b128 per operand (LDS laid out so each lane's 16 K-offsets are two
// contiguous 8-element runs, row stride padded to 80B for bank spread).
// GEMM variants are template-instantiated so there is no runtime branching.
// ---------------------------------------------------------------------------

#define B_   128
#define L_   512
#define D_   128
#define NB_  2
#define MAXLEN_ 512

typedef __attribute__((ext_vector_type(16))) __bf16          v16bf;
typedef __attribute__((ext_vector_type(8)))  float           v8f;
typedef __attribute__((ext_vector_type(8)))  unsigned short  ushort8;
typedef __attribute__((ext_vector_type(4)))  unsigned short  ushort4v;

__device__ __forceinline__ unsigned short f2bf(float f) {
    union { float f; unsigned int u; } x; x.f = f;
    unsigned int u = x.u + 0x7FFFu + ((x.u >> 16) & 1u);   // round-to-nearest-even
    return (unsigned short)(u >> 16);
}

__device__ __forceinline__ float waveSum(float v) {
    #pragma unroll
    for (int o = 16; o > 0; o >>= 1) v += __shfl_xor(v, o, 32);
    return v;
}
__device__ __forceinline__ float waveMax(float v) {
    #pragma unroll
    for (int o = 16; o > 0; o >>= 1) v = fmaxf(v, __shfl_xor(v, o, 32));
    return v;
}

// ---------------------------------------------------------------------------
// WMMA GEMM: C[b] = act(alpha * A[b] @ B[b] + bias) + res[b]
//   BMODE 0: B element (k,n) = Bm[k*ldb + n]   (row-major K x N)
//   BMODE 1: B element (k,n) = Bm[n*ldb + k]   (X @ W^T with W[N,K])
//   ACT   0: none, 1: relu, 2: silu
// Block: 256 threads = 8 waves; wave w owns the 16x16 tile at
//   (blockIdx.y*128 + w*16, blockIdx.x*16).  K % 32 == 0, M % 128 == 0,
//   N % 16 == 0 (true for all calls here).
// ---------------------------------------------------------------------------
template<int BMODE, int ACT, bool HASB, bool HASR>
__global__ __launch_bounds__(256) void wmma_gemm_kernel(
    const float* __restrict__ A, const float* __restrict__ Bm,
    const float* __restrict__ bias, const float* __restrict__ res,
    float* __restrict__ C,
    int K, int lda, int ldb, int ldc,
    long long sA, long long sB, long long sC, float alpha)
{
    constexpr int LR = 40;                       // LDS row stride (ushorts); 80B
    __shared__ unsigned short sAt[128 * LR];     // A tile: [row][k] bf16 bits
    __shared__ unsigned short sBt[16 * LR];      // B tile: [n][k]  bf16 bits

    const int tid  = threadIdx.x;
    const int wave = tid >> 5;
    const int lane = tid & 31;
    const int n0   = blockIdx.x * 16;
    const int mblk = blockIdx.y * 128;
    const int bb   = blockIdx.z;

    const float* Ab = A  + (long long)bb * sA;
    const float* Bb = Bm + (long long)bb * sB;
    float*       Cb = C  + (long long)bb * sC;
    const float* Rb = HASR ? res + (long long)bb * sC : nullptr;

    const int grp  = lane >> 4;                  // 0: lanes 0-15, 1: lanes 16-31
    const int nn   = lane & 15;
    const int arow = wave * 16 + nn;

    v8f acc = {};
    for (int kc = 0; kc < K; kc += 32) {
        // ---- stage A 128x32 chunk: float4 global loads, ushort4 LDS stores
        #pragma unroll
        for (int q = tid; q < 1024; q += 256) {  // 1024 quads of 4 floats
            int rr = q >> 3, kq = (q & 7) * 4;
            const float4 f = *(const float4*)&Ab[(long long)(mblk + rr) * lda + kc + kq];
            ushort4v o; o.x = f2bf(f.x); o.y = f2bf(f.y); o.z = f2bf(f.z); o.w = f2bf(f.w);
            *(ushort4v*)&sAt[rr * LR + kq] = o;
        }
        // ---- stage B 32x16 chunk transposed to [n][k]
        if (BMODE == 1) {                        // W[N,K]: contiguous along k
            if (tid < 128) {                     // 128 quads
                int cn = tid >> 3, kq = (tid & 7) * 4;
                const float4 f = *(const float4*)&Bb[(long long)(n0 + cn) * ldb + kc + kq];
                ushort4v o; o.x = f2bf(f.x); o.y = f2bf(f.y); o.z = f2bf(f.z); o.w = f2bf(f.w);
                *(ushort4v*)&sBt[cn * LR + kq] = o;
            }
        } else {                                 // row-major [K,N]: transpose on store
            #pragma unroll
            for (int e = tid; e < 512; e += 256) {
                int kk = e >> 4, cn = e & 15;
                sBt[cn * LR + kk] = f2bf(Bb[(long long)(kc + kk) * ldb + (n0 + cn)]);
            }
        }
        __syncthreads();

        // ---- fragments: two ds_load_b128 each (CDNA5 16-bit layout: lane's
        //      K-offsets are runs [grp*8, +8) and [16+grp*8, +8))
        union Frag { v16bf v; ushort8 h[2]; } af, bfr;
        const int ao = arow * LR + grp * 8;
        af.h[0]  = *(const ushort8*)&sAt[ao];
        af.h[1]  = *(const ushort8*)&sAt[ao + 16];
        const int bo = nn * LR + grp * 8;
        bfr.h[0] = *(const ushort8*)&sBt[bo];
        bfr.h[1] = *(const ushort8*)&sBt[bo + 16];

        acc = __builtin_amdgcn_wmma_f32_16x16x32_bf16(
                  false, af.v, false, bfr.v, (short)0, acc, false, false);
        __syncthreads();
    }

    // ---- epilogue: C/D layout: VGPR v holds M = v + 8*grp, N = lane&15
    const int m0 = mblk + wave * 16;
    const int nc = n0 + nn;
    float bv = 0.f;
    if (HASB) bv = bias[nc];
    #pragma unroll
    for (int v = 0; v < 8; ++v) {
        int mr = m0 + v + 8 * grp;
        float val = acc[v] * alpha;
        if (HASB) val += bv;
        if (ACT == 1) val = fmaxf(val, 0.f);
        else if (ACT == 2) val = val / (1.f + __expf(-val));   // SiLU
        long long o = (long long)mr * ldc + nc;
        if (HASR) val += Rb[o];
        Cb[o] = val;
    }
}

// ---------------------------------------------------------------------------
// LayerNorm over D=128, one wave per row (in-place safe).
// ---------------------------------------------------------------------------
__global__ __launch_bounds__(256) void ln_kernel(
    const float* __restrict__ x, float* __restrict__ y,
    const float* __restrict__ s, const float* __restrict__ b)
{
    int row  = blockIdx.x * 8 + (threadIdx.x >> 5);
    int lane = threadIdx.x & 31;
    const float4 v = *(const float4*)(x + (long long)row * D_ + lane * 4);
    const float4 sv = *(const float4*)(s + lane * 4);
    const float4 bvv = *(const float4*)(b + lane * 4);
    float mean = waveSum(v.x + v.y + v.z + v.w) * (1.f / D_);
    float dx = v.x - mean, dy = v.y - mean, dz = v.z - mean, dw = v.w - mean;
    float var = waveSum(dx * dx + dy * dy + dz * dz + dw * dw) * (1.f / D_);
    float inv = rsqrtf(var + 1e-8f);
    float4 o;
    o.x = dx * inv * sv.x + bvv.x;
    o.y = dy * inv * sv.y + bvv.y;
    o.z = dz * inv * sv.z + bvv.z;
    o.w = dw * inv * sv.w + bvv.w;
    *(float4*)(y + (long long)row * D_ + lane * 4) = o;
}

// ---------------------------------------------------------------------------
// Causal softmax over L=512 keys, in place; one wave per (b,q) row.
// ---------------------------------------------------------------------------
__global__ __launch_bounds__(256) void softmax_kernel(float* __restrict__ attn)
{
    int row  = blockIdx.x * 8 + (threadIdx.x >> 5);
    int lane = threadIdx.x & 31;
    int qi   = row & (L_ - 1);
    float* a = attn + (long long)row * L_;
    float x[16], mx = -1e30f;
    #pragma unroll
    for (int j = 0; j < 16; ++j) {
        int kk = j * 32 + lane;
        float t = (kk <= qi) ? a[kk] : -1e9f;
        x[j] = t; mx = fmaxf(mx, t);
    }
    mx = waveMax(mx);
    float sm = 0.f;
    #pragma unroll
    for (int j = 0; j < 16; ++j) { x[j] = __expf(x[j] - mx); sm += x[j]; }
    float r = 1.f / waveSum(sm);
    #pragma unroll
    for (int j = 0; j < 16; ++j) a[j * 32 + lane] = x[j] * r;
}

// ---------------------------------------------------------------------------
// CAPE: per (b,q) row: G = 1-sigmoid(attn); P = suffix cumsum; clamp,
// floor/ceil gather from E, lerp, mean over k; seqs += mean (fused).
// One wave per row; lane t owns contiguous chunk [16t, 16t+16).
// ---------------------------------------------------------------------------
__global__ __launch_bounds__(256) void cape_kernel(
    const float* __restrict__ attn, const float* __restrict__ E,
    float* __restrict__ seqs)
{
    int row  = blockIdx.x * 8 + (threadIdx.x >> 5);
    int lane = threadIdx.x & 31;
    const float* ar = attn + (long long)row * L_;
    const float* Er = E    + (long long)row * MAXLEN_;

    float g[16];
    #pragma unroll
    for (int j = 0; j < 16; j += 4) {
        const float4 a = *(const float4*)&ar[lane * 16 + j];
        g[j + 0] = 1.f - 1.f / (1.f + __expf(-a.x));
        g[j + 1] = 1.f - 1.f / (1.f + __expf(-a.y));
        g[j + 2] = 1.f - 1.f / (1.f + __expf(-a.z));
        g[j + 3] = 1.f - 1.f / (1.f + __expf(-a.w));
    }
    float pj[16], run = 0.f;
    #pragma unroll
    for (int j = 15; j >= 0; --j) { run += g[j]; pj[j] = run; }
    float pre = run;
    #pragma unroll
    for (int d = 1; d < 32; d <<= 1) {
        float t = __shfl_up(pre, d, 32);
        if (lane >= d) pre += t;
    }
    float totalAll = __shfl(pre, 31, 32);
    float sufExcl  = totalAll - pre;            // chunks strictly after mine

    float sum = 0.f;
    #pragma unroll
    for (int j = 0; j < 16; ++j) {
        float P  = fminf(pj[j] + sufExcl, (float)(MAXLEN_ - 1));
        float Pf = floorf(P), Pc = ceilf(P);
        float Ef = Er[(int)Pf], Ec = Er[(int)Pc];
        float fr = P - Pf;
        sum += fr * Ec + (1.f - fr) * Ef;
    }
    float mean = waveSum(sum) * (1.f / L_);
    float* sr = seqs + (long long)row * D_ + lane * 4;
    float4 sv = *(const float4*)sr;
    sv.x += mean; sv.y += mean; sv.z += mean; sv.w += mean;
    *(float4*)sr = sv;
}

// ---------------------------------------------------------------------------
// seqs = item_emb[log_seqs] * sqrt(D)
// ---------------------------------------------------------------------------
__global__ __launch_bounds__(256) void gather_kernel(
    const float* __restrict__ item_emb, const int* __restrict__ idx,
    float* __restrict__ out)
{
    long long t = (long long)blockIdx.x * 256 + threadIdx.x;  // quad index
    long long r = t >> 5;                                     // row (32 quads/row)
    int d = (int)(t & 31) * 4;
    int it = idx[r];
    float4 e = *(const float4*)(item_emb + (long long)it * D_ + d);
    const float sc = 11.313708498984761f;                     // sqrt(128)
    e.x *= sc; e.y *= sc; e.z *= sc; e.w *= sc;
    *(float4*)(out + r * D_ + d) = e;
}

// ---------------------------------------------------------------------------
// pos/neg logits: dot(feats[row], item_emb[pos/neg[row]]); one wave per row.
// ---------------------------------------------------------------------------
__global__ __launch_bounds__(256) void logits_kernel(
    const float* __restrict__ feats, const float* __restrict__ item_emb,
    const int* __restrict__ pos, const int* __restrict__ neg,
    float* __restrict__ out)
{
    int row  = blockIdx.x * 8 + (threadIdx.x >> 5);
    int lane = threadIdx.x & 31;
    const float4 f  = *(const float4*)(feats + (long long)row * D_ + lane * 4);
    const float4 pe = *(const float4*)(item_emb + (long long)pos[row] * D_ + lane * 4);
    const float4 ne = *(const float4*)(item_emb + (long long)neg[row] * D_ + lane * 4);
    float ps = waveSum(f.x * pe.x + f.y * pe.y + f.z * pe.z + f.w * pe.w);
    float ns = waveSum(f.x * ne.x + f.y * ne.y + f.z * ne.z + f.w * ne.w);
    if (lane == 0) { out[row] = ps; out[B_ * L_ + row] = ns; }
}

// ---------------------------------------------------------------------------
extern "C" void kernel_launch(void* const* d_in, const int* in_sizes, int n_in,
                              void* d_out, int out_size, void* d_ws, size_t ws_size,
                              hipStream_t stream) {
    (void)in_sizes; (void)n_in; (void)out_size; (void)ws_size;

    const float* item_emb = (const float*)d_in[0];
    const float* pos_emb  = (const float*)d_in[1];
    const float* pre_w    = (const float*)d_in[2];
    const float* pre_b    = (const float*)d_in[3];
    const float* ln1_s    = (const float*)d_in[4];
    const float* ln1_b    = (const float*)d_in[5];
    const float* in_w     = (const float*)d_in[6];
    const float* in_b     = (const float*)d_in[7];
    const float* out_w    = (const float*)d_in[8];
    const float* out_b    = (const float*)d_in[9];
    const float* ln2_s    = (const float*)d_in[10];
    const float* ln2_b    = (const float*)d_in[11];
    const float* c1_w     = (const float*)d_in[12];
    const float* c1_b     = (const float*)d_in[13];
    const float* c2_w     = (const float*)d_in[14];
    const float* c2_b     = (const float*)d_in[15];
    const float* lnf_s    = (const float*)d_in[16];
    const float* lnf_b    = (const float*)d_in[17];
    // d_in[18] user_ids: unused by the reference computation
    const int* log_seqs   = (const int*)d_in[19];
    const int* pos_seqs   = (const int*)d_in[20];
    const int* neg_seqs   = (const int*)d_in[21];
    float* out = (float*)d_out;

    const long long BLD = (long long)B_ * L_ * D_;      //  8,388,608 f32
    const long long BLL = (long long)B_ * L_ * L_;      // 33,554,432 f32
    const long long LD  = (long long)L_ * D_;
    float* w     = (float*)d_ws;
    float* seqsA = w;
    float* seqsB = w + BLD;
    float* Qln   = w + 2 * BLD;
    float* qbuf  = w + 3 * BLD;
    float* kbuf  = w + 4 * BLD;
    float* vbuf  = w + 5 * BLD;
    float* t1    = w + 6 * BLD;                         // attn_out / qp / ffn-h
    float* attn  = w + 7 * BLD;
    float* Ebuf  = attn + BLL;

    const int ROWS = B_ * L_;           // 65536
    const dim3 blk(256);
    const int rowBlks = ROWS / 8;       // wave-per-row kernels

    // seqs = item_emb[log_seqs] * sqrt(D)
    gather_kernel<<<(int)(BLD / 4 / 256), blk, 0, stream>>>(item_emb, log_seqs, seqsA);

    float* S  = seqsA;
    float* Sn = seqsB;
    const float invSqrtD = 0.0883883476483184f;         // 1/sqrt(128)

    for (int i = 0; i < NB_; ++i) {
        const float* wq = in_w + (long long)i * 3 * D_ * D_;
        const float* wk = wq + D_ * D_;
        const float* wv = wk + D_ * D_;
        const float* bq = in_b + (long long)i * 3 * D_;

        // Q = LN(seqs)
        ln_kernel<<<rowBlks, blk, 0, stream>>>(S, Qln, ln1_s + i * D_, ln1_b + i * D_);

        // q = Q@wq^T+bq, k = seqs@wk^T+bk, v = seqs@wv^T+bv
        wmma_gemm_kernel<1,0,true,false><<<dim3(D_/16, ROWS/128, 1), blk, 0, stream>>>(
            Qln, wq, bq, nullptr, qbuf, D_, D_, D_, D_, 0, 0, 0, 1.f);
        wmma_gemm_kernel<1,0,true,false><<<dim3(D_/16, ROWS/128, 1), blk, 0, stream>>>(
            S, wk, bq + D_, nullptr, kbuf, D_, D_, D_, D_, 0, 0, 0, 1.f);
        wmma_gemm_kernel<1,0,true,false><<<dim3(D_/16, ROWS/128, 1), blk, 0, stream>>>(
            S, wv, bq + 2 * D_, nullptr, vbuf, D_, D_, D_, D_, 0, 0, 0, 1.f);

        // scores[b] = q[b] @ k[b]^T * (1/sqrt(D))   (batched over B)
        wmma_gemm_kernel<1,0,false,false><<<dim3(L_/16, L_/128, B_), blk, 0, stream>>>(
            qbuf, kbuf, nullptr, nullptr, attn, D_, D_, D_, L_, LD, LD, BLL / B_, invSqrtD);

        // causal softmax in place
        softmax_kernel<<<rowBlks, blk, 0, stream>>>(attn);

        // t1[b] = attn[b] @ v[b]
        wmma_gemm_kernel<0,0,false,false><<<dim3(D_/16, L_/128, B_), blk, 0, stream>>>(
            attn, vbuf, nullptr, nullptr, t1, L_, L_, D_, D_, BLL / B_, LD, LD, 1.f);

        // seqs = Q + t1 @ out_w^T + out_b
        wmma_gemm_kernel<1,0,true,true><<<dim3(D_/16, ROWS/128, 1), blk, 0, stream>>>(
            t1, out_w + (long long)i * D_ * D_, out_b + i * D_, Qln, S,
            D_, D_, D_, D_, 0, 0, 0, 1.f);

        // qp = silu(Q @ pre_w^T + pre_b)   (reuse t1)
        wmma_gemm_kernel<1,2,true,false><<<dim3(D_/16, ROWS/128, 1), blk, 0, stream>>>(
            Qln, pre_w, pre_b, nullptr, t1, D_, D_, D_, D_, 0, 0, 0, 1.f);

        // E = qp @ pos_emb   ([B*L, 512])
        wmma_gemm_kernel<0,0,false,false><<<dim3(MAXLEN_/16, ROWS/128, 1), blk, 0, stream>>>(
            t1, pos_emb, nullptr, nullptr, Ebuf, D_, D_, MAXLEN_, MAXLEN_, 0, 0, 0, 1.f);

        // CAPE: seqs += mean_k lerp(E, reversed-cumsum(1-sigmoid(attn)))
        cape_kernel<<<rowBlks, blk, 0, stream>>>(attn, Ebuf, S);

        // seqs = LN2(seqs) in place
        ln_kernel<<<rowBlks, blk, 0, stream>>>(S, S, ln2_s + i * D_, ln2_b + i * D_);

        // h = relu(seqs @ c1^T + b1); seqs' = seqs + h @ c2^T + b2
        wmma_gemm_kernel<1,1,true,false><<<dim3(D_/16, ROWS/128, 1), blk, 0, stream>>>(
            S, c1_w + (long long)i * D_ * D_, c1_b + i * D_, nullptr, t1,
            D_, D_, D_, D_, 0, 0, 0, 1.f);
        wmma_gemm_kernel<1,0,true,true><<<dim3(D_/16, ROWS/128, 1), blk, 0, stream>>>(
            t1, c2_w + (long long)i * D_ * D_, c2_b + i * D_, S, Sn,
            D_, D_, D_, D_, 0, 0, 0, 1.f);

        float* tmp = S; S = Sn; Sn = tmp;   // ping-pong
    }

    // feats = LN(seqs, lnf)
    ln_kernel<<<rowBlks, blk, 0, stream>>>(S, Qln, lnf_s, lnf_b);

    // pos/neg logits
    logits_kernel<<<rowBlks, blk, 0, stream>>>(Qln, item_emb, pos_seqs, neg_seqs, out);
}
